// DistributedMoE_74105365725765
// MI455X (gfx1250) — compile-verified, matrix operand order checked by reading, not run on hardware
//
#include <hip/hip_runtime.h>
#include <hip/hip_bf16.h>

typedef _Float16 h16;
typedef __attribute__((ext_vector_type(16))) _Float16 v16h;
typedef __attribute__((ext_vector_type(8)))  _Float16 v8h;
typedef __attribute__((ext_vector_type(8)))  float    v8f;

#define NEXPERT 6
#define NCLASS  10
#define CAPF    192.0f

static __device__ inline v16h cat8(v8h lo, v8h hi) {
  return __builtin_shufflevector(lo, hi, 0,1,2,3,4,5,6,7,8,9,10,11,12,13,14,15);
}

// ---------------------------------------------------------------------------
// f32 -> f16 convert
// ---------------------------------------------------------------------------
__global__ void cvt_f32_f16(const float* __restrict__ in, h16* __restrict__ out, int n) {
  int i = blockIdx.x * blockDim.x + threadIdx.x;
  if (i < n) out[i] = (h16)in[i];
}

// ---------------------------------------------------------------------------
// Pack conv weights: wp[co][Kpad] f16 = w[co][k] * s(co), k = cin*9+kh*3+kw,
// zero-padded to Kpad. beta[co] = cb*s + bb. s = g/sqrt(1+eps).
// ---------------------------------------------------------------------------
__global__ void packw_k(const float* __restrict__ w, const float* __restrict__ cb,
                        const float* __restrict__ bg, const float* __restrict__ bb,
                        h16* __restrict__ wp, float* __restrict__ beta,
                        int Cout, int Ktot, int Kpad, int hasCb) {
  const float inv = rsqrtf(1.0f + 1e-5f);
  int idx = blockIdx.x * blockDim.x + threadIdx.x;
  int total = Cout * Kpad;
  if (idx < total) {
    int co = idx / Kpad; int k = idx - co * Kpad;
    float s = bg[co] * inv;
    float v = (k < Ktot) ? w[co * Ktot + k] * s : 0.f;
    wp[idx] = (h16)v;
  }
  if (idx < Cout) {
    float s = bg[idx] * inv;
    beta[idx] = (hasCb ? cb[idx] : 0.f) * s + bb[idx];
  }
}

// ---------------------------------------------------------------------------
// Pack FC weights: w16[n][K] f16 (n zero-padded to Npad), b32[n] zero-padded.
// ---------------------------------------------------------------------------
__global__ void packfc_k(const float* __restrict__ w, const float* __restrict__ bias,
                         h16* __restrict__ w16, float* __restrict__ b32,
                         int Nact, int Npad, int K) {
  int idx = blockIdx.x * blockDim.x + threadIdx.x;
  int total = Npad * K;
  if (idx < total) {
    int n = idx / K; int k = idx - n * K;
    w16[idx] = (n < Nact) ? (h16)w[n * K + k] : (h16)0.f;
  }
  if (idx < Npad) b32[idx] = (idx < Nact) ? bias[idx] : 0.f;
}

// ---------------------------------------------------------------------------
// Implicit-GEMM 3x3 SAME conv, WMMA f32_16x16x32_f16.
// A fragments: direct global b128 loads from packed, BN-folded f16 weights.
// B fragments: branch-free clamped im2col staged through LDS, reused for
// NCO cout tiles -> NCO*NPIX WMMAs per staged chunk.
// One wave computes (16*NCO) couts x 64 pixels. out = relu(acc + beta).
// ---------------------------------------------------------------------------
template<int Cin, int H, int W, int Cout, int NCO>
__global__ __launch_bounds__(32) void conv3x3_wmma(
    const h16*  __restrict__ in,    // [B,Cin,H,W] f16
    const h16*  __restrict__ wp,    // [Cout,Kpad] f16, BN-scaled
    const float* __restrict__ beta, // [Cout] folded bias
    h16* __restrict__ out)          // [B,Cout,H,W] f16
{
  constexpr int HW   = H * W;
  constexpr int Ktot = Cin * 9;
  constexpr int Kpad = (Ktot + 31) & ~31;
  constexpr int NPIX = 4;

  __shared__ __align__(16) h16 lB[NPIX][16 * 32];

  const int tid      = threadIdx.x;
  const int pixBase  = blockIdx.x * (16 * NPIX);
  const int coutBase = blockIdx.y * (16 * NCO);

  const int mr = tid & 15;          // fragment row (M for A, N for B)
  const int kb = (tid >> 4) * 8;    // K sub-block per lane half

  const h16* arow[NCO];
  #pragma unroll
  for (int c = 0; c < NCO; ++c)
    arow[c] = wp + (size_t)(coutBase + c * 16 + mr) * Kpad;

  // staging assignment: this thread fills lB[t][sn*32 + sh .. +15]
  const int sn = tid >> 1;
  const int sh = (tid & 1) * 16;

  v8f acc[NCO][NPIX] = {};

  for (int k0 = 0; k0 < Ktot; k0 += 32) {
    if (k0 + 32 < Ktot) __builtin_prefetch(arow[0] + k0 + 32, 0, 0);

    // ---- stage B tiles, branch-free ----
    #pragma unroll
    for (int t = 0; t < NPIX; ++t) {
      int p  = pixBase + t * 16 + sn;
      int bi = p / HW;  int rp = p - bi * HW;
      int oh = rp / W;  int ow = rp - oh * W;
      const h16* base = in + (size_t)bi * Cin * HW;
      h16 vals[16];
      bool inb[16];
      #pragma unroll
      for (int i = 0; i < 16; ++i) {
        int k = k0 + sh + i;
        int cin = k / 9; int r = k - cin * 9;
        int kh = r / 3;  int kw = r - kh * 3;
        int ih = oh + kh - 1, iw = ow + kw - 1;
        bool ok = (k < Ktot) & ((unsigned)ih < (unsigned)H) & ((unsigned)iw < (unsigned)W);
        inb[i] = ok;
        int idx = (cin * H + (ok ? ih : 0)) * W + (ok ? iw : 0);  // clamped, always valid
        vals[i] = base[idx];                                      // unconditional load
      }
      v8h lo, hi;
      #pragma unroll
      for (int i = 0; i < 8; ++i) {
        lo[i] = inb[i]     ? vals[i]     : (h16)0.f;
        hi[i] = inb[i + 8] ? vals[i + 8] : (h16)0.f;
      }
      *(v8h*)&lB[t][sn * 32 + sh]     = lo;
      *(v8h*)&lB[t][sn * 32 + sh + 8] = hi;
    }

    // ---- A fragments: direct from packed global weights ----
    v16h a[NCO];
    #pragma unroll
    for (int c = 0; c < NCO; ++c)
      a[c] = cat8(*(const v8h*)(arow[c] + k0 + kb),
                  *(const v8h*)(arow[c] + k0 + 16 + kb));

    __syncthreads();
    #pragma unroll
    for (int t = 0; t < NPIX; ++t) {
      v16h b = cat8(*(const v8h*)&lB[t][mr * 32 + kb],
                    *(const v8h*)&lB[t][mr * 32 + 16 + kb]);
      #pragma unroll
      for (int c = 0; c < NCO; ++c)
        acc[c][t] = __builtin_amdgcn_wmma_f32_16x16x32_f16(false, a[c], false, b,
                                                           (short)0, acc[c][t], false, false);
    }
    __syncthreads();
  }

  // ---- epilogue ----
  const int nn = tid & 15;
  const int mh = (tid >> 4) * 8;
  #pragma unroll
  for (int t = 0; t < NPIX; ++t) {
    int p  = pixBase + t * 16 + nn;
    int bi = p / HW;  int rp = p - bi * HW;
    int oh = rp / W;  int ow = rp - oh * W;
    #pragma unroll
    for (int c = 0; c < NCO; ++c) {
      #pragma unroll
      for (int r = 0; r < 8; ++r) {
        int co = coutBase + c * 16 + mh + r;
        float v = fmaxf(acc[c][t][r] + beta[co], 0.f);
        out[((size_t)(bi * Cout + co) * H + oh) * W + ow] = (h16)v;
      }
    }
  }
}

// ---------------------------------------------------------------------------
// LDS-free WMMA GEMM: out[m][n] = act(in16[m][K] . w16[n][K] + bias[n]).
// Both fragments direct b128 register loads. Dual f32/f16 outputs (nullable).
// ---------------------------------------------------------------------------
__global__ __launch_bounds__(32) void gemm_wmma(
    const h16* __restrict__ in16, const h16* __restrict__ w16,
    const float* __restrict__ bias,
    float* __restrict__ out32, h16* __restrict__ out16,
    int K, int Nact, int ldo, int relu)
{
  const int tid   = threadIdx.x;
  const int mBase = blockIdx.x * 16;
  const int nBase = blockIdx.y * 16;
  const int mr = tid & 15;
  const int kb = (tid >> 4) * 8;
  const h16* ar = in16 + (size_t)(mBase + mr) * K;
  const h16* br = w16 + (size_t)(nBase + mr) * K;
  v8f acc = {};

  for (int k0 = 0; k0 < K; k0 += 32) {
    v16h a = cat8(*(const v8h*)(ar + k0 + kb), *(const v8h*)(ar + k0 + 16 + kb));
    v16h b = cat8(*(const v8h*)(br + k0 + kb), *(const v8h*)(br + k0 + 16 + kb));
    acc = __builtin_amdgcn_wmma_f32_16x16x32_f16(false, a, false, b, (short)0, acc, false, false);
  }

  const int nn = tid & 15;
  const int mh = (tid >> 4) * 8;
  if (nBase + nn < Nact) {
    float bv = bias[nBase + nn];
    #pragma unroll
    for (int r = 0; r < 8; ++r) {
      float v = acc[r] + bv;
      if (relu) v = fmaxf(v, 0.f);
      size_t o = (size_t)(mBase + mh + r) * ldo + nBase + nn;
      if (out32) out32[o] = v;
      if (out16) out16[o] = (h16)v;
    }
  }
}

// ---------------------------------------------------------------------------
// 2x2 maxpool stride 2 on f16 [BC,H,W] -> [BC,H/2,W/2]
// ---------------------------------------------------------------------------
__global__ void maxpool2_k(const h16* __restrict__ in, h16* __restrict__ out,
                           int BC, int H, int W) {
  int OH = H >> 1, OW = W >> 1;
  int n = BC * OH * OW;
  int idx = blockIdx.x * blockDim.x + threadIdx.x;
  if (idx >= n) return;
  int bc = idx / (OH * OW); int r = idx - bc * OH * OW;
  int oh = r / OW; int ow = r - oh * OW;
  const h16* p = in + ((size_t)bc * H + oh * 2) * W + ow * 2;
  float m = fmaxf(fmaxf((float)p[0], (float)p[1]),
                  fmaxf((float)p[W], (float)p[W + 1]));
  out[idx] = (h16)m;
}

// ---------------------------------------------------------------------------
// Trunk head: maxpool2 (32->16) then avg to 2x2, fused. out [B,128] f16.
// ---------------------------------------------------------------------------
__global__ void trunkpool_k(const h16* __restrict__ in, h16* __restrict__ out, int B) {
  int idx = blockIdx.x * blockDim.x + threadIdx.x;
  if (idx >= B * 128) return;
  int b = idx >> 7; int r = idx & 127;
  int c = r >> 2; int ij = r & 3; int i = ij >> 1; int j = ij & 1;
  const h16* p = in + ((size_t)(b * 32 + c)) * 32 * 32;
  float acc = 0.f;
  for (int ph = 0; ph < 8; ++ph)
    for (int pw = 0; pw < 8; ++pw) {
      int r0 = i * 16 + ph * 2, c0 = j * 16 + pw * 2;
      float m = fmaxf(fmaxf((float)p[r0 * 32 + c0],       (float)p[r0 * 32 + c0 + 1]),
                      fmaxf((float)p[(r0 + 1) * 32 + c0], (float)p[(r0 + 1) * 32 + c0 + 1]));
      acc += m;
    }
  out[idx] = (h16)(acc * (1.f / 64.f));
}

// ---------------------------------------------------------------------------
// Global average pool: in [n,HW] f16 -> out [n] f16
// ---------------------------------------------------------------------------
__global__ void gap_k(const h16* __restrict__ in, h16* __restrict__ out, int n, int HW) {
  int idx = blockIdx.x * blockDim.x + threadIdx.x;
  if (idx >= n) return;
  const h16* p = in + (size_t)idx * HW;
  float acc = 0.f;
  for (int i = 0; i < HW; ++i) acc += (float)p[i];
  out[idx] = (h16)(acc / (float)HW);
}

// ---------------------------------------------------------------------------
// Per (b,e): conf = sum p*log(p+1e-12); balanced = .7*gate+.3*conf+boost-2*ema
// ---------------------------------------------------------------------------
__global__ void balance_k(const float* __restrict__ logits_e,
                          const float* __restrict__ gate,
                          const float* __restrict__ ema,
                          float* __restrict__ balanced, int B) {
  int idx = blockIdx.x * blockDim.x + threadIdx.x;
  if (idx >= B * NEXPERT) return;
  int b = idx / NEXPERT; int e = idx - b * NEXPERT;
  const float* lg = logits_e + (size_t)(b * NEXPERT + e) * NCLASS;
  float mx = lg[0];
  for (int c = 1; c < NCLASS; ++c) mx = fmaxf(mx, lg[c]);
  float se = 0.f;
  for (int c = 0; c < NCLASS; ++c) se += expf(lg[c] - mx);
  float lse = mx + logf(se);
  float conf = 0.f;
  for (int c = 0; c < NCLASS; ++c) {
    float p = expf(lg[c] - lse);
    conf += p * logf(p + 1e-12f);
  }
  float em = ema[e];
  float boost = (em < 0.05f) ? (0.05f - em) * 10.f : 0.f;
  balanced[idx] = 0.7f * gate[idx] + 0.3f * conf + boost - 2.0f * em;
}

// ---------------------------------------------------------------------------
// Sequential greedy capacity-constrained dispatch (exact scan semantics).
// ---------------------------------------------------------------------------
__global__ void dispatch_k(const float* __restrict__ balanced, int* __restrict__ chosen, int B) {
  if (threadIdx.x != 0 || blockIdx.x != 0) return;
  float loads[NEXPERT];
  for (int e = 0; e < NEXPERT; ++e) loads[e] = 0.f;
  for (int b = 0; b < B; ++b) {
    const float* r = balanced + b * NEXPERT;
    int i0 = 0;
    for (int e = 1; e < NEXPERT; ++e) if (r[e] > r[i0]) i0 = e;      // first max
    int i1 = -1;
    for (int e = 0; e < NEXPERT; ++e)
      if (e != i0 && (i1 < 0 || r[e] > r[i1])) i1 = e;               // second (first-tie)
    float l0 = loads[i0], l1 = loads[i1];
    bool f0 = l0 < CAPF, f1 = l1 < CAPF;
    int ch;
    if (f0)       ch = i0;
    else if (f1)  ch = i1;
    else          ch = (l0 <= l1) ? i0 : i1;                          // argmin, first-tie
    loads[ch] += 1.f;
    chosen[b] = ch;
  }
}

// ---------------------------------------------------------------------------
// Finalize: logits_final[b] = logits_e[b,chosen[b]]; D one-hot 0/1 floats.
// ---------------------------------------------------------------------------
__global__ void finalize_k(const float* __restrict__ logits_e, const int* __restrict__ chosen,
                           float* __restrict__ out_logits, float* __restrict__ out_D, int B) {
  int b = blockIdx.x * blockDim.x + threadIdx.x;
  if (b >= B) return;
  int ch = chosen[b];
  const float* src = logits_e + (size_t)(b * NEXPERT + ch) * NCLASS;
  for (int c = 0; c < NCLASS; ++c) out_logits[b * NCLASS + c] = src[c];
  for (int e = 0; e < NEXPERT; ++e) out_D[b * NEXPERT + e] = (e == ch) ? 1.f : 0.f;
}

// ---------------------------------------------------------------------------
extern "C" void kernel_launch(void* const* d_in, const int* in_sizes, int n_in,
                              void* d_out, int out_size, void* d_ws, size_t ws_size,
                              hipStream_t stream) {
  (void)in_sizes; (void)n_in; (void)out_size; (void)ws_size;
  const int B = 1024;

  const float* x     = (const float*)d_in[0];
  const float* t_c1w = (const float*)d_in[1];
  const float* t_b1g = (const float*)d_in[2];
  const float* t_b1b = (const float*)d_in[3];
  const float* t_c2w = (const float*)d_in[4];
  const float* t_b2g = (const float*)d_in[5];
  const float* t_b2b = (const float*)d_in[6];
  const float* t_fcw = (const float*)d_in[7];
  const float* t_fcb = (const float*)d_in[8];
  const float* g1w   = (const float*)d_in[9];
  const float* g1b   = (const float*)d_in[10];
  const float* g2w   = (const float*)d_in[11];
  const float* g2b   = (const float*)d_in[12];
  const float* e_c1w = (const float*)d_in[13];
  const float* e_c1b = (const float*)d_in[14];
  const float* e_b1g = (const float*)d_in[15];
  const float* e_b1b = (const float*)d_in[16];
  const float* e_c2w = (const float*)d_in[17];
  const float* e_c2b = (const float*)d_in[18];
  const float* e_b2g = (const float*)d_in[19];
  const float* e_b2b = (const float*)d_in[20];
  const float* e_c3w = (const float*)d_in[21];
  const float* e_c3b = (const float*)d_in[22];
  const float* e_b3g = (const float*)d_in[23];
  const float* e_b3b = (const float*)d_in[24];
  const float* e_c4w = (const float*)d_in[25];
  const float* e_c4b = (const float*)d_in[26];
  const float* e_b4g = (const float*)d_in[27];
  const float* e_b4b = (const float*)d_in[28];
  const float* e_c5w = (const float*)d_in[29];
  const float* e_c5b = (const float*)d_in[30];
  const float* e_b5g = (const float*)d_in[31];
  const float* e_b5b = (const float*)d_in[32];
  const float* e_fw  = (const float*)d_in[33];
  const float* e_fb  = (const float*)d_in[34];
  const float* e_cw  = (const float*)d_in[35];
  const float* e_cb  = (const float*)d_in[36];
  const float* ema   = (const float*)d_in[37];

  // ---- workspace layout ----
  char* ws = (char*)d_ws;
  size_t off = 0;
  auto alloc = [&](size_t bytes) { size_t o = off; off = (off + bytes + 255) & ~(size_t)255; return o; };
  size_t o_x16  = alloc((size_t)B * 3   * 32 * 32 * 2);
  size_t o_a1   = alloc((size_t)B * 32  * 32 * 32 * 2);   // conv1 out / trunk c1 out
  size_t o_a2   = alloc((size_t)B * 64  * 16 * 16 * 2);   // conv2 out  } aT aliases a2+a3
  size_t o_a3   = alloc((size_t)B * 64  * 16 * 16 * 2);   // conv3 out  }
  size_t o_a1p  = alloc((size_t)B * 32  * 16 * 16 * 2);
  size_t o_a3p  = alloc((size_t)B * 64  * 8  * 8  * 2);
  size_t o_a4   = alloc((size_t)B * 128 * 8  * 8  * 2);
  size_t o_a5   = alloc((size_t)B * 128 * 8  * 8  * 2);
  size_t o_wp   = alloc((size_t)128 * 1152 * 2);          // packed conv weights (max)
  size_t o_wb   = alloc((size_t)128 * 4);                 // folded conv bias
  size_t o_fw   = alloc((size_t)128 * 128 * 2);           // packed fc weights (max)
  size_t o_fb   = alloc((size_t)128 * 4);                 // padded fc bias
  size_t o_h16  = alloc((size_t)B * 128 * 2);
  size_t o_rf   = alloc((size_t)B * 64  * 2);
  size_t o_g    = alloc((size_t)B * 32  * 2);
  size_t o_gate = alloc((size_t)B * NEXPERT * 4);
  size_t o_feat = alloc((size_t)B * 128 * 2);
  size_t o_f    = alloc((size_t)B * 128 * 2);
  size_t o_le   = alloc((size_t)B * NEXPERT * NCLASS * 4);
  size_t o_ch   = alloc((size_t)B * 4);

  h16*   x16   = (h16*)(ws + o_x16);
  h16*   a1    = (h16*)(ws + o_a1);
  h16*   a2    = (h16*)(ws + o_a2);
  h16*   a3    = (h16*)(ws + o_a3);
  h16*   aT    = (h16*)(ws + o_a2);   // trunk c2 out (dead before a2/a3 used)
  h16*   a1p   = (h16*)(ws + o_a1p);
  h16*   a3p   = (h16*)(ws + o_a3p);
  h16*   a4    = (h16*)(ws + o_a4);
  h16*   a5    = (h16*)(ws + o_a5);
  h16*   wp    = (h16*)(ws + o_wp);
  float* wbeta = (float*)(ws + o_wb);
  h16*   fw16  = (h16*)(ws + o_fw);
  float* fb32  = (float*)(ws + o_fb);
  h16*   h16b  = (h16*)(ws + o_h16);
  h16*   rf16  = (h16*)(ws + o_rf);
  h16*   g16   = (h16*)(ws + o_g);
  float* gate  = (float*)(ws + o_gate);
  h16*   feat16= (h16*)(ws + o_feat);
  h16*   fbuf16= (h16*)(ws + o_f);
  float* lge   = (float*)(ws + o_le);
  int*   chose = (int*)(ws + o_ch);

  float* out_logits   = (float*)d_out;                     // [B,10]
  float* out_balanced = (float*)d_out + B * NCLASS;        // [B,6]
  float* out_D        = out_balanced + B * NEXPERT;        // [B,6]

  auto packw = [&](const float* w, const float* cb, const float* bg, const float* bb,
                   int Cout, int Cin, int hasCb) {
    int Ktot = Cin * 9, Kpad = (Ktot + 31) & ~31;
    int total = Cout * Kpad;
    packw_k<<<dim3((total + 255) / 256), 256, 0, stream>>>(w, cb, bg, bb, wp, wbeta,
                                                           Cout, Ktot, Kpad, hasCb);
  };
  auto packfc = [&](const float* w, const float* bias, int Nact, int Npad, int K) {
    int total = Npad * K;
    packfc_k<<<dim3((total + 255) / 256), 256, 0, stream>>>(w, bias, fw16, fb32, Nact, Npad, K);
  };

  // ---- 0. input to f16 ----
  {
    int n = B * 3 * 32 * 32;
    cvt_f32_f16<<<dim3(n / 256), dim3(256), 0, stream>>>(x, x16, n);
  }

  // ---- 1. routing trunk ----
  packw(t_c1w, nullptr, t_b1g, t_b1b, 32, 3, 0);
  conv3x3_wmma<3, 32, 32, 32, 2><<<dim3(16384, 1), 32, 0, stream>>>(x16, wp, wbeta, a1);
  packw(t_c2w, nullptr, t_b2g, t_b2b, 32, 32, 0);
  conv3x3_wmma<32, 32, 32, 32, 2><<<dim3(16384, 1), 32, 0, stream>>>(a1, wp, wbeta, aT);
  trunkpool_k<<<dim3(B * 128 / 256), 256, 0, stream>>>(aT, h16b, B);
  packfc(t_fcw, t_fcb, 64, 64, 128);
  gemm_wmma<<<dim3(B / 16, 4), 32, 0, stream>>>(h16b, fw16, fb32, nullptr, rf16, 128, 64, 64, 1);
  packfc(g1w, g1b, 32, 32, 64);
  gemm_wmma<<<dim3(B / 16, 2), 32, 0, stream>>>(rf16, fw16, fb32, nullptr, g16, 64, 32, 32, 1);
  packfc(g2w, g2b, NEXPERT, 16, 32);
  gemm_wmma<<<dim3(B / 16, 1), 32, 0, stream>>>(g16, fw16, fb32, gate, nullptr, 32, NEXPERT, NEXPERT, 0);

  // ---- 2. experts (sequential, buffers reused) ----
  for (int e = 0; e < NEXPERT; ++e) {
    packw(e_c1w + (size_t)e * 32 * 3 * 9, e_c1b + e * 32, e_b1g + e * 32, e_b1b + e * 32, 32, 3, 1);
    conv3x3_wmma<3, 32, 32, 32, 2><<<dim3(16384, 1), 32, 0, stream>>>(x16, wp, wbeta, a1);
    maxpool2_k<<<dim3(B * 32 * 16 * 16 / 256), 256, 0, stream>>>(a1, a1p, B * 32, 32, 32);

    packw(e_c2w + (size_t)e * 64 * 32 * 9, e_c2b + e * 64, e_b2g + e * 64, e_b2b + e * 64, 64, 32, 1);
    conv3x3_wmma<32, 16, 16, 64, 4><<<dim3(4096, 1), 32, 0, stream>>>(a1p, wp, wbeta, a2);

    packw(e_c3w + (size_t)e * 64 * 64 * 9, e_c3b + e * 64, e_b3g + e * 64, e_b3b + e * 64, 64, 64, 1);
    conv3x3_wmma<64, 16, 16, 64, 4><<<dim3(4096, 1), 32, 0, stream>>>(a2, wp, wbeta, a3);
    maxpool2_k<<<dim3(B * 64 * 8 * 8 / 256), 256, 0, stream>>>(a3, a3p, B * 64, 16, 16);

    packw(e_c4w + (size_t)e * 128 * 64 * 9, e_c4b + e * 128, e_b4g + e * 128, e_b4b + e * 128, 128, 64, 1);
    conv3x3_wmma<64, 8, 8, 128, 4><<<dim3(1024, 2), 32, 0, stream>>>(a3p, wp, wbeta, a4);

    packw(e_c5w + (size_t)e * 128 * 128 * 9, e_c5b + e * 128, e_b5g + e * 128, e_b5b + e * 128, 128, 128, 1);
    conv3x3_wmma<128, 8, 8, 128, 4><<<dim3(1024, 2), 32, 0, stream>>>(a4, wp, wbeta, a5);

    gap_k<<<dim3(B * 128 / 256), 256, 0, stream>>>(a5, feat16, B * 128, 64);

    packfc(e_fw + (size_t)e * 128 * 128, e_fb + e * 128, 128, 128, 128);
    gemm_wmma<<<dim3(B / 16, 8), 32, 0, stream>>>(feat16, fw16, fb32, nullptr, fbuf16, 128, 128, 128, 1);

    packfc(e_cw + (size_t)e * NCLASS * 128, e_cb + e * NCLASS, NCLASS, 16, 128);
    gemm_wmma<<<dim3(B / 16, 1), 32, 0, stream>>>(fbuf16, fw16, fb32, lge + e * NCLASS, nullptr,
                                                  128, NCLASS, NEXPERT * NCLASS, 0);
  }

  // ---- 3. routing scores, dispatch, combine ----
  {
    int n = B * NEXPERT;
    balance_k<<<dim3((n + 255) / 256), 256, 0, stream>>>(lge, gate, ema, out_balanced, B);
  }
  dispatch_k<<<1, 32, 0, stream>>>(out_balanced, chose, B);
  finalize_k<<<dim3(B / 256), 256, 0, stream>>>(lge, chose, out_logits, out_D, B);
}
